// SelectiveMoELayer_69432441307314
// MI455X (gfx1250) — compile-verified, matrix operand order checked by reading, not run on hardware
//
#include <hip/hip_runtime.h>

// ---------------- fixed problem shape (from reference setup_inputs) ----------
#define T_TOK 4096      // B*S = 2*2048 tokens
#define HDIM  1024
#define FDIM  2816
#define NE    16
#define TOPK  8
#define NROWS (T_TOK * TOPK)   // 32768 gathered (token,expert) rows

// GEMM tiling
#define BM    128              // rows per block (8 m-tiles of 16)
#define KS    128              // K super-chunk staged in LDS
#define LDSLD 136              // padded LDS row stride in bf16 (68 dwords -> no bank conflicts)

// CDNA5 async global->LDS staging (ASYNCcnt path), guarded so the file still
// compiles on toolchains without the builtins.
#if defined(__has_builtin)
#if __has_builtin(__builtin_amdgcn_global_load_async_to_lds_b128)
#define HAVE_ASYNC_LDS 1
#endif
#if __has_builtin(__builtin_amdgcn_s_wait_asynccnt)
#define HAVE_WAIT_ASYNC 1
#endif
#endif

typedef __attribute__((ext_vector_type(4))) int v4i;
typedef __attribute__((address_space(1))) v4i* gas_v4i;   // global addrspace ptr
typedef __attribute__((address_space(3))) v4i* las_v4i;   // LDS addrspace ptr

__device__ __forceinline__ void async_wait_all() {
#if defined(HAVE_ASYNC_LDS)
#if defined(HAVE_WAIT_ASYNC)
    __builtin_amdgcn_s_wait_asynccnt(0);
#else
    asm volatile("s_wait_asynccnt 0" ::: "memory");
#endif
#endif
}

__device__ __forceinline__ void stage_b128(const uint4* g, uint4* l) {
#if defined(HAVE_ASYNC_LDS)
    __builtin_amdgcn_global_load_async_to_lds_b128(
        (gas_v4i)(v4i*)g, (las_v4i)(v4i*)l, 0 /*offset*/, 0 /*cpol*/);
#else
    *l = *g;
#endif
}

typedef __attribute__((ext_vector_type(16))) __bf16 v16bf;
typedef __attribute__((ext_vector_type(8)))  __bf16 v8bf;
typedef __attribute__((ext_vector_type(4)))  __bf16 v4bf;
typedef __attribute__((ext_vector_type(8)))  float  v8f;

// ---------------------------------------------------------------------------
// Fragment loaders. CDNA5 16-bit A-matrix 16x32 layout (ISA 7.12.2):
//   lanes 0-15 : row M = lane,      per-lane elems = K[0..7], K[16..23]
//   lanes 16-31: row M = lane-16,   per-lane elems = K[8..15], K[24..31]
// B 32x16 mirrors this with N = lane&15. C/D f32: M = vgpr + 8*(lane>>4),
// N = lane&15.
// ---------------------------------------------------------------------------
__device__ __forceinline__ v16bf load_a_frag_lds(const __bf16* p /* row + k0 + 8*khalf */) {
    v8bf lo = *(const v8bf*)(p);        // K + 0..7
    v8bf hi = *(const v8bf*)(p + 16);   // K + 16..23
    v16bf r;
#pragma unroll
    for (int j = 0; j < 8; ++j) { r[j] = lo[j]; r[8 + j] = hi[j]; }
    return r;
}

__device__ __forceinline__ v16bf load_b_frag_f32(const float* p /* base + k0 + 8*khalf */) {
    const float4* q0 = (const float4*)(p);
    const float4* q1 = (const float4*)(p + 16);
    float4 a = q0[0], b = q0[1];   // K + 0..7
    float4 c = q1[0], d = q1[1];   // K + 16..23
    v16bf r;
    r[0]  = (__bf16)a.x; r[1]  = (__bf16)a.y; r[2]  = (__bf16)a.z; r[3]  = (__bf16)a.w;
    r[4]  = (__bf16)b.x; r[5]  = (__bf16)b.y; r[6]  = (__bf16)b.z; r[7]  = (__bf16)b.w;
    r[8]  = (__bf16)c.x; r[9]  = (__bf16)c.y; r[10] = (__bf16)c.z; r[11] = (__bf16)c.w;
    r[12] = (__bf16)d.x; r[13] = (__bf16)d.y; r[14] = (__bf16)d.z; r[15] = (__bf16)d.w;
    return r;
}

// ---------------------------------------------------------------------------
// 0) x fp32 -> bf16 (activations staged once; weights converted in-register)
// ---------------------------------------------------------------------------
__global__ __launch_bounds__(256) void cvt_x_kernel(const float* __restrict__ x,
                                                    __bf16* __restrict__ xb) {
    size_t i = ((size_t)blockIdx.x * 256 + threadIdx.x) * 4;   // T*H divisible
    float4 v = *(const float4*)(x + i);
    v4bf o;
    o[0] = (__bf16)v.x; o[1] = (__bf16)v.y; o[2] = (__bf16)v.z; o[3] = (__bf16)v.w;
    *(v4bf*)(xb + i) = o;
}

// ---------------------------------------------------------------------------
// 1) router logits [T,E], fp32 exact (decides routing -> keep full precision)
// ---------------------------------------------------------------------------
__global__ __launch_bounds__(256) void router_kernel(const float* __restrict__ x,
                                                     const float* __restrict__ rw,
                                                     float* __restrict__ logits) {
    int tid = blockIdx.x * 256 + threadIdx.x;   // T*E threads
    int t = tid >> 4, e = tid & 15;
    if (t >= T_TOK) return;
    const float4* xa = (const float4*)(x + (size_t)t * HDIM);
    const float4* wa = (const float4*)(rw + (size_t)e * HDIM);
    float acc = 0.f;
#pragma unroll 4
    for (int i = 0; i < HDIM / 4; ++i) {
        float4 a = xa[i], b = wa[i];
        acc += a.x * b.x + a.y * b.y + a.z * b.z + a.w * b.w;
    }
    logits[t * NE + e] = acc;
}

__global__ __launch_bounds__(64) void zero_counts_kernel(int* __restrict__ counts) {
    if (threadIdx.x < NE) counts[threadIdx.x] = 0;
}

// ---------------------------------------------------------------------------
// 2) per-token top-8 (desc, lowest index on ties like lax.top_k) + softmax
// ---------------------------------------------------------------------------
__global__ __launch_bounds__(256) void topk_kernel(const float* __restrict__ logits,
                                                   int* __restrict__ tk_idx,
                                                   float* __restrict__ tk_w,
                                                   int* __restrict__ counts) {
    int t = blockIdx.x * 256 + threadIdx.x;
    if (t >= T_TOK) return;
    float v[NE];
#pragma unroll
    for (int e = 0; e < NE; ++e) v[e] = logits[t * NE + e];
    int idx[TOPK]; float val[TOPK];
    unsigned mask = 0;
#pragma unroll
    for (int k = 0; k < TOPK; ++k) {
        int best = 0; float bv = -1e30f;
#pragma unroll
        for (int e = 0; e < NE; ++e)
            if (!((mask >> e) & 1u) && v[e] > bv) { bv = v[e]; best = e; }
        mask |= 1u << best; idx[k] = best; val[k] = bv;
    }
    float mx = val[0], s = 0.f, w[TOPK];
#pragma unroll
    for (int k = 0; k < TOPK; ++k) { w[k] = __expf(val[k] - mx); s += w[k]; }
    float inv = 1.f / s;
#pragma unroll
    for (int k = 0; k < TOPK; ++k) {
        tk_idx[t * TOPK + k] = idx[k];
        tk_w[t * TOPK + k]   = w[k] * inv;
        atomicAdd(&counts[idx[k]], 1);
    }
}

// 3) exclusive scan over 16 counts (single thread, trivial)
__global__ void scan_kernel(const int* __restrict__ counts,
                            int* __restrict__ offsets, int* __restrict__ cursors) {
    if (threadIdx.x == 0) {
        int o = 0;
        for (int e = 0; e < NE; ++e) { offsets[e] = o; cursors[e] = o; o += counts[e]; }
    }
}

// 4) assign each (token,k) a slot in its expert's contiguous row range.
//    Slot order is nondeterministic but the final output is order-invariant.
__global__ __launch_bounds__(256) void assign_kernel(const int* __restrict__ tk_idx,
                                                     int* __restrict__ cursors,
                                                     int* __restrict__ expert_tokens,
                                                     int* __restrict__ row_of) {
    int t = blockIdx.x * 256 + threadIdx.x;
    if (t >= T_TOK) return;
#pragma unroll
    for (int k = 0; k < TOPK; ++k) {
        int e = tk_idx[t * TOPK + k];
        int pos = atomicAdd(&cursors[e], 1);
        expert_tokens[pos] = t;
        row_of[t * TOPK + k] = pos;
    }
}

// ---------------------------------------------------------------------------
// 5) fused gate/up GEMM + SwiGLU.  Block = 128 gathered rows x 128 F cols.
//    8 waves; each wave owns 16 F-cols and 8 row-tiles for gate AND up, so
//    every converted B fragment feeds 8 WMMAs (1 cvt_pk per WMMA).
//    A staged in LDS via async global->LDS (ASYNCcnt) in 128-wide K
//    super-chunks; rows padded to 136 bf16 so half-wave ds_load_b128 hit
//    distinct banks.
// ---------------------------------------------------------------------------
__global__ __launch_bounds__(256) void gateup_kernel(const __bf16* __restrict__ xb,
                                                     const float* __restrict__ gate_w,
                                                     const float* __restrict__ up_w,
                                                     const int* __restrict__ offsets,
                                                     const int* __restrict__ counts,
                                                     const int* __restrict__ expert_tokens,
                                                     __bf16* __restrict__ h_out) {
    const int e   = blockIdx.z;
    const int cnt = counts[e];
    const int tile = blockIdx.y;
    if (tile * BM >= cnt) return;
    const int off      = offsets[e];
    const int slot_end = off + cnt;
    const int row_base = off + tile * BM;
    const int f0       = blockIdx.x * 128;

    __shared__ __bf16 Alds[BM * LDSLD];   // 34 KB padded A tile
    __shared__ int    toks[BM];

    const int tid = threadIdx.x;
    for (int i = tid; i < BM; i += 256) {
        int slot = row_base + i;
        if (slot >= slot_end) slot = slot_end - 1;   // clamp padded rows
        toks[i] = expert_tokens[slot];
    }

    const int wave  = tid >> 5;
    const int lane  = tid & 31;
    const int n     = lane & 15;
    const int khalf = lane >> 4;
    const int fw    = f0 + wave * 16;
    const int mrow  = lane & 15;

    const float* gb = gate_w + ((size_t)e * FDIM + fw + n) * HDIM + 8 * khalf;
    const float* ub = up_w   + ((size_t)e * FDIM + fw + n) * HDIM + 8 * khalf;

    v8f cg[8] = {};
    v8f cu[8] = {};

    for (int ks = 0; ks < HDIM / KS; ++ks) {
        __syncthreads();   // all waves done reading previous tile
        // stage BM x KS bf16 (rows padded): 128 rows x 16 x b128
        for (int i = tid; i < BM * (KS / 8); i += 256) {
            int r = i >> 4, c = i & 15;
            stage_b128(((const uint4*)(xb + (size_t)toks[r] * HDIM + ks * KS)) + c,
                       (uint4*)(Alds + (size_t)r * LDSLD + c * 8));
        }
        async_wait_all();
        __syncthreads();
#pragma unroll
        for (int k0 = 0; k0 < KS; k0 += 32) {
            v16bf bg = load_b_frag_f32(gb + ks * KS + k0);
            v16bf bu = load_b_frag_f32(ub + ks * KS + k0);
#pragma unroll
            for (int m = 0; m < 8; ++m) {
                const __bf16* ap = Alds + (size_t)(m * 16 + mrow) * LDSLD + k0 + 8 * khalf;
                v16bf a = load_a_frag_lds(ap);
                cg[m] = __builtin_amdgcn_wmma_f32_16x16x32_bf16(false, a, false, bg, (short)0, cg[m], false, false);
                cu[m] = __builtin_amdgcn_wmma_f32_16x16x32_bf16(false, a, false, bu, (short)0, cu[m], false, false);
            }
        }
    }
    // SwiGLU epilogue: h = silu(g) * u, store bf16 rows (guard padded rows)
#pragma unroll
    for (int m = 0; m < 8; ++m) {
#pragma unroll
        for (int r = 0; r < 8; ++r) {
            int row = row_base + m * 16 + r + 8 * khalf;
            if (row < slot_end) {
                float g = cg[m][r], u = cu[m][r];
                float h = (g / (1.f + __expf(-g))) * u;
                h_out[(size_t)row * FDIM + fw + n] = (__bf16)h;
            }
        }
    }
}

// ---------------------------------------------------------------------------
// 6) down GEMM.  Block = 128 rows x 128 H cols; K=F=2816 in 22 LDS-staged
//    super-chunks of 128.  Each converted B fragment feeds 8 WMMAs.
// ---------------------------------------------------------------------------
__global__ __launch_bounds__(256) void down_kernel(const __bf16* __restrict__ h_in,
                                                   const float* __restrict__ down_w,
                                                   const int* __restrict__ offsets,
                                                   const int* __restrict__ counts,
                                                   float* __restrict__ out_rows) {
    const int e   = blockIdx.z;
    const int cnt = counts[e];
    const int tile = blockIdx.y;
    if (tile * BM >= cnt) return;
    const int off      = offsets[e];
    const int slot_end = off + cnt;
    const int row_base = off + tile * BM;
    const int h0       = blockIdx.x * 128;

    __shared__ __bf16 Alds[BM * LDSLD];   // 34 KB padded A tile

    const int tid   = threadIdx.x;
    const int wave  = tid >> 5;
    const int lane  = tid & 31;
    const int n     = lane & 15;
    const int khalf = lane >> 4;
    const int hw    = h0 + wave * 16;
    const int mrow  = lane & 15;

    const float* db = down_w + ((size_t)e * HDIM + hw + n) * FDIM + 8 * khalf;

    v8f c[8] = {};

    for (int ks = 0; ks < FDIM / KS; ++ks) {
        __syncthreads();
        for (int i = tid; i < BM * (KS / 8); i += 256) {
            int r = i >> 4, cc = i & 15;
            int slot = row_base + r;
            if (slot >= slot_end) slot = slot_end - 1;
            stage_b128(((const uint4*)(h_in + (size_t)slot * FDIM + ks * KS)) + cc,
                       (uint4*)(Alds + (size_t)r * LDSLD + cc * 8));
        }
        async_wait_all();
        __syncthreads();
#pragma unroll
        for (int k0 = 0; k0 < KS; k0 += 32) {
            v16bf b = load_b_frag_f32(db + ks * KS + k0);
#pragma unroll
            for (int m = 0; m < 8; ++m) {
                const __bf16* ap = Alds + (size_t)(m * 16 + mrow) * LDSLD + k0 + 8 * khalf;
                v16bf a = load_a_frag_lds(ap);
                c[m] = __builtin_amdgcn_wmma_f32_16x16x32_bf16(false, a, false, b, (short)0, c[m], false, false);
            }
        }
    }
#pragma unroll
    for (int m = 0; m < 8; ++m) {
#pragma unroll
        for (int r = 0; r < 8; ++r) {
            int row = row_base + m * 16 + r + 8 * khalf;
            if (row < slot_end)
                out_rows[(size_t)row * HDIM + hw + n] = c[m][r];
        }
    }
}

// ---------------------------------------------------------------------------
// 7) deterministic combine: out[t] = sum_k w[t,k] * out_rows[row_of[t,k]]
// ---------------------------------------------------------------------------
__global__ __launch_bounds__(256) void combine_kernel(const float* __restrict__ out_rows,
                                                      const int* __restrict__ row_of,
                                                      const float* __restrict__ tk_w,
                                                      float* __restrict__ out) {
    const int t   = blockIdx.x;
    const int tid = threadIdx.x;           // 256 threads, one float4 each
    float4 acc = {0.f, 0.f, 0.f, 0.f};
#pragma unroll
    for (int k = 0; k < TOPK; ++k) {
        int   row = row_of[t * TOPK + k];
        float w   = tk_w[t * TOPK + k];
        float4 v = ((const float4*)(out_rows + (size_t)row * HDIM))[tid];
        acc.x += w * v.x; acc.y += w * v.y; acc.z += w * v.z; acc.w += w * v.w;
    }
    ((float4*)(out + (size_t)t * HDIM))[tid] = acc;
}

// ---------------------------------------------------------------------------
extern "C" void kernel_launch(void* const* d_in, const int* in_sizes, int n_in,
                              void* d_out, int out_size, void* d_ws, size_t ws_size,
                              hipStream_t stream) {
    const float* x       = (const float*)d_in[0];   // [T, H]
    const float* rw      = (const float*)d_in[1];   // [E, H]
    const float* gate_w  = (const float*)d_in[2];   // [E, F, H]
    const float* up_w    = (const float*)d_in[3];   // [E, F, H]
    const float* down_w  = (const float*)d_in[4];   // [E, H, F]
    float*       out     = (float*)d_out;           // [T, H]

    // workspace carve-up (256B aligned regions)
    char* ws = (char*)d_ws;
    size_t o = 0;
    auto carve = [&](size_t bytes) -> char* {
        char* p = ws + o;
        o += (bytes + 255) & ~(size_t)255;
        return p;
    };
    float*  logits        = (float*)carve((size_t)T_TOK * NE * 4);
    int*    tk_idx        = (int*)  carve((size_t)T_TOK * TOPK * 4);
    float*  tk_w          = (float*)carve((size_t)T_TOK * TOPK * 4);
    int*    row_of        = (int*)  carve((size_t)T_TOK * TOPK * 4);
    int*    counts        = (int*)  carve(NE * 4);
    int*    offsets       = (int*)  carve(NE * 4);
    int*    cursors       = (int*)  carve(NE * 4);
    int*    expert_tokens = (int*)  carve((size_t)NROWS * 4);
    __bf16* xb            = (__bf16*)carve((size_t)T_TOK * HDIM * 2);
    __bf16* h_buf         = (__bf16*)carve((size_t)NROWS * FDIM * 2);
    float*  out_rows      = (float*)carve((size_t)NROWS * HDIM * 4);

    // 0) activations -> bf16
    cvt_x_kernel<<<(T_TOK * HDIM) / (256 * 4), 256, 0, stream>>>(x, xb);
    // 1) router
    zero_counts_kernel<<<1, 64, 0, stream>>>(counts);
    router_kernel<<<(T_TOK * NE) / 256, 256, 0, stream>>>(x, rw, logits);
    // 2) top-8 + softmax + counts
    topk_kernel<<<T_TOK / 256, 256, 0, stream>>>(logits, tk_idx, tk_w, counts);
    // 3) scan, 4) slot assignment
    scan_kernel<<<1, 32, 0, stream>>>(counts, offsets, cursors);
    assign_kernel<<<T_TOK / 256, 256, 0, stream>>>(tk_idx, cursors, expert_tokens, row_of);
    // 5) gate/up + SwiGLU  (grid: 22 F-tiles x up-to-32 row-blocks x 16 experts)
    gateup_kernel<<<dim3(FDIM / 128, T_TOK / BM, NE), 256, 0, stream>>>(
        xb, gate_w, up_w, offsets, counts, expert_tokens, h_buf);
    // 6) down projection (grid: 8 H-tiles x row-blocks x experts)
    down_kernel<<<dim3(HDIM / 128, T_TOK / BM, NE), 256, 0, stream>>>(
        h_buf, down_w, offsets, counts, out_rows);
    // 7) weighted combine (deterministic, no float atomics)
    combine_kernel<<<T_TOK, 256, 0, stream>>>(out_rows, row_of, tk_w, out);
}
